// CausalSelfAttention_15582141350064
// MI455X (gfx1250) — matrix operand order repaired and checked by reading, not compiled, not run on hardware
//
#include <hip/hip_runtime.h>

// ---------------------------------------------------------------------------
// CDNA5 (gfx1250) causal self-attention, bf16 WMMA pipeline.
// B=8, S=2048, E=512, H=8, DH=64
// ---------------------------------------------------------------------------

typedef __bf16 bf16;
typedef __attribute__((ext_vector_type(16))) __bf16 v16bf;
typedef __attribute__((ext_vector_type(8)))  float  v8f;
typedef __attribute__((ext_vector_type(4)))  unsigned int u32x4;

union FragBF {
    v16bf v;
    u32x4 q[2];
};

__device__ __forceinline__ v8f wmma_bf16(v16bf a, v16bf b, v8f c) {
    // D = A(16x32 bf16) * B(32x16 bf16) + C(16x16 f32)
    return __builtin_amdgcn_wmma_f32_16x16x32_bf16(
        /*neg_a=*/false, a, /*neg_b=*/false, b,
        /*c_mod=*/(short)0, c, /*reuse_a=*/false, /*reuse_b=*/false);
}

#define Bb 8
#define Ss 2048
#define Ee 512
#define Hh 8
#define DH 64
#define MROWS (Bb * Ss)          // 16384
#define NEGBIG (-1.0e30f)

// ---------------------------------------------------------------------------
// fp32 -> bf16 convert
// ---------------------------------------------------------------------------
__global__ void cvt_f32_bf16(const float* __restrict__ src, bf16* __restrict__ dst, int n) {
    int i = blockIdx.x * blockDim.x + threadIdx.x;
    if (i < n) dst[i] = (bf16)src[i];
}

// Wt[n][k] = (bf16) W[k][n]   (W is [K][N] row-major)
__global__ void transpose_cvt(const float* __restrict__ W, bf16* __restrict__ Wt, int K, int N) {
    int i = blockIdx.x * blockDim.x + threadIdx.x;
    if (i < K * N) {
        int n = i / K;
        int kk = i - n * K;
        Wt[(size_t)n * K + kk] = (bf16)W[(size_t)kk * N + n];
    }
}

// ---------------------------------------------------------------------------
// QKV GEMM:  [16384 x 512] * [512 x 1536] + bias, scatter into q/k/vt layouts
//   q,k : [B,H,S,DH] bf16      vt : [B,H,DH,S] bf16
// One wave computes a 16x64 tile (4 wmma accumulators).
// ---------------------------------------------------------------------------
__global__ void __launch_bounds__(256)
gemm_qkv(const bf16* __restrict__ X, const bf16* __restrict__ Wt,
         const float* __restrict__ bias,
         bf16* __restrict__ qo, bf16* __restrict__ ko, bf16* __restrict__ vto) {
    const int lane = threadIdx.x & 31;
    const int wv   = threadIdx.x >> 5;
    const int gw   = blockIdx.x * 8 + wv;
    const int nT   = (3 * Ee) / 64;          // 24
    const int mt   = gw / nT;
    const int n64  = gw - mt * nT;
    const int half = lane >> 4;
    const int l16  = lane & 15;

    v8f acc[4] = {};
    const bf16* arow = X + (size_t)(mt * 16 + l16) * Ee;

    for (int kk = 0; kk < Ee; kk += 32) {
        FragBF a;
        a.q[0] = *(const u32x4*)(arow + kk + 8 * half);
        a.q[1] = *(const u32x4*)(arow + kk + 16 + 8 * half);
#pragma unroll
        for (int j = 0; j < 4; ++j) {
            const bf16* brow = Wt + (size_t)(n64 * 64 + j * 16 + l16) * Ee;
            FragBF bfrag;
            bfrag.q[0] = *(const u32x4*)(brow + kk + 8 * half);
            bfrag.q[1] = *(const u32x4*)(brow + kk + 16 + 8 * half);
            acc[j] = wmma_bf16(a.v, bfrag.v, acc[j]);
        }
    }

    // epilogue: n64*64 block maps to a single (t, h) pair (64 == DH)
    const int ncol0 = n64 * 64;
    const int t   = ncol0 >> 9;           // 0=q 1=k 2=v
    const int h   = (ncol0 & 511) >> 6;
#pragma unroll
    for (int j = 0; j < 4; ++j) {
        const int d = j * 16 + l16;
        const float bia = bias[ncol0 + j * 16 + l16];
#pragma unroll
        for (int r = 0; r < 8; ++r) {
            const int m  = mt * 16 + r + 8 * half;   // global row = b*S + s
            const int bb = m >> 11;
            const int s  = m & (Ss - 1);
            const bf16 val = (bf16)(acc[j][r] + bia);
            if (t == 0)
                qo[((size_t)(bb * Hh + h) * Ss + s) * DH + d] = val;
            else if (t == 1)
                ko[((size_t)(bb * Hh + h) * Ss + s) * DH + d] = val;
            else
                vto[((size_t)(bb * Hh + h) * DH + d) * Ss + s] = val;
        }
    }
}

// ---------------------------------------------------------------------------
// Flash attention: one wave owns a 32-row q block (2 row-tiles), online
// softmax over 32-wide k chunks. K/V fragments are loaded once per chunk and
// reused by both row-tiles (halves L2 traffic vs 16-row waves). Probs are
// restaged via a per-wave 32x32 LDS tile into WMMA A-fragments.
// ---------------------------------------------------------------------------
__global__ void __launch_bounds__(128)
attn_kernel(const bf16* __restrict__ qg, const bf16* __restrict__ kg,
            const bf16* __restrict__ vtg, const float* __restrict__ rel,
            bf16* __restrict__ aout) {
    __shared__ __align__(16) bf16 lds[4][32 * 32];

    const int lane = threadIdx.x & 31;
    const int wv   = threadIdx.x >> 5;
    const int w    = blockIdx.x * 4 + wv;     // (b*H + h)*64 + qt
    const int qt   = w & 63;
    const int bh   = w >> 6;
    const int h    = bh & (Hh - 1);
    const int half = lane >> 4;
    const int l16  = lane & 15;
    const int qbase = qt * 32;

    const bf16* qp = qg + (size_t)bh * Ss * DH;
    const bf16* kp = kg + (size_t)bh * Ss * DH;
    const bf16* vp = vtg + (size_t)bh * DH * Ss;
    const float* rp = rel + (size_t)h * Ss * Ss;
    const float scale = 0.125f;               // DH^-0.5

    // Q A-fragments: 2 row-tiles x 2 K-halves, loaded once
    FragBF qa[2][2];
#pragma unroll
    for (int i = 0; i < 2; ++i) {
        const bf16* qrow = qp + (size_t)(qbase + i * 16 + l16) * DH;
        qa[i][0].q[0] = *(const u32x4*)(qrow + 8 * half);
        qa[i][0].q[1] = *(const u32x4*)(qrow + 16 + 8 * half);
        qa[i][1].q[0] = *(const u32x4*)(qrow + 32 + 8 * half);
        qa[i][1].q[1] = *(const u32x4*)(qrow + 48 + 8 * half);
    }

    float mrow[2][8], lrow[2][8];
    v8f O[2][4] = {};
#pragma unroll
    for (int i = 0; i < 2; ++i)
#pragma unroll
        for (int r = 0; r < 8; ++r) { mrow[i][r] = NEGBIG; lrow[i][r] = 0.0f; }

    const int kend = qbase + 32;
    for (int kb = 0; kb < kend; kb += 32) {
        // ---- K^T B-fragments for this chunk (shared by both row-tiles)
        const bf16* k0 = kp + (size_t)(kb + l16) * DH;
        const bf16* k1 = kp + (size_t)(kb + 16 + l16) * DH;
        FragBF b00, b01, b10, b11;
        b00.q[0] = *(const u32x4*)(k0 + 8 * half);
        b00.q[1] = *(const u32x4*)(k0 + 16 + 8 * half);
        b01.q[0] = *(const u32x4*)(k0 + 32 + 8 * half);
        b01.q[1] = *(const u32x4*)(k0 + 48 + 8 * half);
        b10.q[0] = *(const u32x4*)(k1 + 8 * half);
        b10.q[1] = *(const u32x4*)(k1 + 16 + 8 * half);
        b11.q[0] = *(const u32x4*)(k1 + 32 + 8 * half);
        b11.q[1] = *(const u32x4*)(k1 + 48 + 8 * half);

        bf16* pt = lds[wv];

#pragma unroll
        for (int i = 0; i < 2; ++i) {
            // ---- scores for row-tile i
            v8f s0 = {}, s1 = {};
            s0 = wmma_bf16(qa[i][0].v, b00.v, s0);
            s0 = wmma_bf16(qa[i][1].v, b01.v, s0);
            s1 = wmma_bf16(qa[i][0].v, b10.v, s1);
            s1 = wmma_bf16(qa[i][1].v, b11.v, s1);

            // ---- rel_pos + scale + causal mask + online softmax
            float alpha[8];
#pragma unroll
            for (int r = 0; r < 8; ++r) {
                const int qr  = qbase + i * 16 + r + 8 * half;
                const int kc0 = kb + l16;
                const int kc1 = kb + 16 + l16;
                float sc0 = s0[r] * scale + rp[(size_t)qr * Ss + kc0];
                float sc1 = s1[r] * scale + rp[(size_t)qr * Ss + kc1];
                if (kc0 > qr) sc0 = NEGBIG;
                if (kc1 > qr) sc1 = NEGBIG;
                float mx = fmaxf(sc0, sc1);
                mx = fmaxf(mx, __shfl_xor(mx, 1));
                mx = fmaxf(mx, __shfl_xor(mx, 2));
                mx = fmaxf(mx, __shfl_xor(mx, 4));
                mx = fmaxf(mx, __shfl_xor(mx, 8));
                const float mnew = fmaxf(mrow[i][r], mx);
                const float a  = __expf(mrow[i][r] - mnew);
                const float e0 = __expf(sc0 - mnew);
                const float e1 = __expf(sc1 - mnew);
                float rs = e0 + e1;
                rs += __shfl_xor(rs, 1);
                rs += __shfl_xor(rs, 2);
                rs += __shfl_xor(rs, 4);
                rs += __shfl_xor(rs, 8);
                lrow[i][r] = lrow[i][r] * a + rs;
                mrow[i][r] = mnew;
                alpha[r] = a;
                // stage probs (C-layout) into the 32x32 bf16 LDS tile
                pt[(i * 16 + r + 8 * half) * 32 + l16]      = (bf16)e0;
                pt[(i * 16 + r + 8 * half) * 32 + 16 + l16] = (bf16)e1;
            }
#pragma unroll
            for (int j = 0; j < 4; ++j)
#pragma unroll
                for (int r = 0; r < 8; ++r) O[i][j][r] *= alpha[r];
        }

        // wave-private tile: per-wave LDS is in-order; block compiler reorder
        asm volatile("s_wait_dscnt 0" ::: "memory");

        FragBF pa[2];
#pragma unroll
        for (int i = 0; i < 2; ++i) {
            const bf16* prow = pt + (i * 16 + l16) * 32;
            pa[i].q[0] = *(const u32x4*)(prow + 8 * half);
            pa[i].q[1] = *(const u32x4*)(prow + 16 + 8 * half);
        }

        // ---- O += P(32x32) @ V(32x64); V fragments shared by both row-tiles
#pragma unroll
        for (int j = 0; j < 4; ++j) {
            const bf16* vrow = vp + (size_t)(j * 16 + l16) * Ss + kb;
            FragBF vb;
            vb.q[0] = *(const u32x4*)(vrow + 8 * half);
            vb.q[1] = *(const u32x4*)(vrow + 16 + 8 * half);
            O[0][j] = wmma_bf16(pa[0].v, vb.v, O[0][j]);
            O[1][j] = wmma_bf16(pa[1].v, vb.v, O[1][j]);
        }
    }

    // ---- normalize and emit attn_out as bf16 [B, S, E] (heads interleaved)
    const int bb = bh >> 3;
#pragma unroll
    for (int i = 0; i < 2; ++i)
#pragma unroll
        for (int j = 0; j < 4; ++j)
#pragma unroll
            for (int r = 0; r < 8; ++r) {
                const int m   = qbase + i * 16 + r + 8 * half;
                const int col = h * DH + j * 16 + l16;
                aout[((size_t)(bb * Ss + m)) * Ee + col] =
                    (bf16)(O[i][j][r] / lrow[i][r]);
            }
}

// ---------------------------------------------------------------------------
// Output projection: [16384 x 512] * [512 x 512] + bias -> fp32
// ---------------------------------------------------------------------------
__global__ void __launch_bounds__(256)
gemm_out(const bf16* __restrict__ A, const bf16* __restrict__ Wt,
         const float* __restrict__ bias, float* __restrict__ out) {
    const int lane = threadIdx.x & 31;
    const int wv   = threadIdx.x >> 5;
    const int gw   = blockIdx.x * 8 + wv;
    const int nT   = Ee / 64;                // 8
    const int mt   = gw / nT;
    const int n64  = gw - mt * nT;
    const int half = lane >> 4;
    const int l16  = lane & 15;

    v8f acc[4] = {};
    const bf16* arow = A + (size_t)(mt * 16 + l16) * Ee;

    for (int kk = 0; kk < Ee; kk += 32) {
        FragBF a;
        a.q[0] = *(const u32x4*)(arow + kk + 8 * half);
        a.q[1] = *(const u32x4*)(arow + kk + 16 + 8 * half);
#pragma unroll
        for (int j = 0; j < 4; ++j) {
            const bf16* brow = Wt + (size_t)(n64 * 64 + j * 16 + l16) * Ee;
            FragBF bfrag;
            bfrag.q[0] = *(const u32x4*)(brow + kk + 8 * half);
            bfrag.q[1] = *(const u32x4*)(brow + kk + 16 + 8 * half);
            acc[j] = wmma_bf16(a.v, bfrag.v, acc[j]);
        }
    }
#pragma unroll
    for (int j = 0; j < 4; ++j) {
        const int n = n64 * 64 + j * 16 + l16;
        const float bia = bias[n];
#pragma unroll
        for (int r = 0; r < 8; ++r) {
            const int m = mt * 16 + r + 8 * half;
            out[(size_t)m * Ee + n] = acc[j][r] + bia;
        }
    }
}

// ---------------------------------------------------------------------------
// Launch
// ---------------------------------------------------------------------------
extern "C" void kernel_launch(void* const* d_in, const int* in_sizes, int n_in,
                              void* d_out, int out_size, void* d_ws, size_t ws_size,
                              hipStream_t stream) {
    (void)in_sizes; (void)n_in; (void)out_size; (void)ws_size;
    const float* x    = (const float*)d_in[0];   // [8,2048,512]
    const float* Wqkv = (const float*)d_in[1];   // [512,1536]
    const float* bqkv = (const float*)d_in[2];   // [1536]
    const float* Wout = (const float*)d_in[3];   // [512,512]
    const float* bout = (const float*)d_in[4];   // [512]
    const float* rel  = (const float*)d_in[5];   // [1,8,2048,2048]
    float* out = (float*)d_out;

    char* ws = (char*)d_ws;
    bf16* xb    = (bf16*)(ws + 0);           // 16,777,216 B
    bf16* wqkvT = (bf16*)(ws + 16777216);    //  1,572,864 B
    bf16* woutT = (bf16*)(ws + 18350080);    //    524,288 B
    bf16* qb    = (bf16*)(ws + 18874368);    // 16,777,216 B
    bf16* kb    = (bf16*)(ws + 35651584);    // 16,777,216 B
    bf16* vtb   = (bf16*)(ws + 52428800);    // 16,777,216 B
    bf16* ab    = (bf16*)(ws + 69206016);    // 16,777,216 B  (end ~86 MB)

    // converts / transposes
    cvt_f32_bf16<<<(MROWS * Ee + 255) / 256, 256, 0, stream>>>(x, xb, MROWS * Ee);
    transpose_cvt<<<(Ee * 3 * Ee + 255) / 256, 256, 0, stream>>>(Wqkv, wqkvT, Ee, 3 * Ee);
    transpose_cvt<<<(Ee * Ee + 255) / 256, 256, 0, stream>>>(Wout, woutT, Ee, Ee);

    // QKV projection: (16384/16)*(1536/64) = 24576 waves, 8 waves/block
    gemm_qkv<<<(MROWS / 16) * ((3 * Ee) / 64) / 8, 256, 0, stream>>>(
        xb, wqkvT, bqkv, qb, kb, vtb);

    // attention: B*H*(S/32) = 4096 waves, 4 waves/block
    attn_kernel<<<(Bb * Hh * (Ss / 32)) / 4, 128, 0, stream>>>(qb, kb, vtb, rel, ab);

    // output projection: (16384/16)*(512/64) = 8192 waves, 8 waves/block
    gemm_out<<<(MROWS / 16) * (Ee / 64) / 8, 256, 0, stream>>>(ab, woutT, bout, out);
}
